// SDPA_31671088841439
// MI455X (gfx1250) — matrix operand order repaired and checked by reading, not compiled, not run on hardware
//
#include <hip/hip_runtime.h>
#include <hip/hip_bf16.h>

typedef __attribute__((ext_vector_type(16))) __bf16 v16bf;
typedef __attribute__((ext_vector_type(8)))  __bf16 v8bf;
typedef __attribute__((ext_vector_type(4)))  __bf16 v4bf;
typedef __attribute__((ext_vector_type(8)))  float  v8f;
typedef __attribute__((ext_vector_type(4)))  unsigned int v4u;
typedef __attribute__((ext_vector_type(8)))  int v8i;
typedef __attribute__((ext_vector_type(4)))  int v4i;

#define N_ROWS 8192
#define HID    1024

// Tensor Data Mover availability / arity selection.
#if __has_builtin(__builtin_amdgcn_tensor_load_to_lds) && \
    __has_builtin(__builtin_amdgcn_s_wait_tensorcnt)
#define USE_TDM 1
#else
#define USE_TDM 0
#endif
#if __has_include(<hip/amd_detail/amd_gfx1250_TDM.h>)
#define TDM_SIX_ARGS 1   // therock-10.0 headers -> clang-23 6-arg builtin
#else
#define TDM_SIX_ARGS 0   // ROCm 7.2 clang-22 5-arg builtin
#endif

// ---------------------------------------------------------------- helpers

__device__ __forceinline__ v16bf cat8(v8bf lo, v8bf hi) {
  return __builtin_shufflevector(lo, hi, 0, 1, 2, 3, 4, 5, 6, 7,
                                         8, 9, 10, 11, 12, 13, 14, 15);
}

// Order-monotonic float <-> uint32 encoding so we can use integer atomicMax
// for a deterministic column max.
__device__ __forceinline__ unsigned enc_f32(float f) {
  unsigned b = __float_as_uint(f);
  return (b & 0x80000000u) ? ~b : (b | 0x80000000u);
}
__device__ __forceinline__ float dec_f32(unsigned u) {
  return __uint_as_float((u & 0x80000000u) ? (u ^ 0x80000000u) : ~u);
}

#if USE_TDM
// Issue one TENSOR_LOAD_TO_LDS for a 128-row x 32-element bf16 tile whose
// rows are K elements apart in global memory, landing contiguously
// (128 x 64B = 8KB) at LDS byte offset `lds_off`. 2D D# per ISA 8.3/8.4:
//   group0: count=1, lds_addr, global_addr[56:0], type=2
//   group1: data_size=1 (2B), tensor_dim0=K, tensor_dim1=128,
//           tile_dim0=32, tile_dim1=128, tensor_dim0_stride=K
// Tracked on TENSORcnt; caller waits with s_wait_tensorcnt.
__device__ __forceinline__ void tdm_load_tile_128x32(
    const __bf16* gsrc, unsigned lds_off, int K)
{
  const unsigned long long ga = (unsigned long long)(size_t)gsrc;
  const unsigned td0 = (unsigned)K;       // tensor dim0 (elements)
  const unsigned td1 = 128u;              // tensor dim1 (rows)
  const unsigned long long st0 = (unsigned long long)(unsigned)K;

  v4u g0;
  g0[0] = 1u;                                           // count=1, user D#
  g0[1] = lds_off;                                      // lds_addr (bytes)
  g0[2] = (unsigned)(ga & 0xFFFFFFFFu);                 // global_addr[31:0]
  g0[3] = (unsigned)((ga >> 32) & 0x01FFFFFFu) |        // global_addr[56:32]
          (2u << 30);                                   // type = 2 ("image")

  v8i g1;
  g1[0] = (int)(1u << 16);                              // data_size=1 -> 2B
  g1[1] = (int)((td0 & 0xFFFFu) << 16);                 // tensor_dim0[15:0]
  g1[2] = (int)(((td0 >> 16) & 0xFFFFu) |               // tensor_dim0[31:16]
                ((td1 & 0xFFFFu) << 16));               // tensor_dim1[15:0]
  g1[3] = (int)(((td1 >> 16) & 0xFFFFu) |               // tensor_dim1[31:16]
                (32u << 16));                           // tile_dim0 = 32
  g1[4] = (int)128u;                                    // tile_dim1=128, d2=0
  g1[5] = (int)(unsigned)(st0 & 0xFFFFFFFFu);           // dim0_stride[31:0]
  g1[6] = (int)(unsigned)((st0 >> 32) & 0xFFFFu);       // dim0_stride[47:32]
  g1[7] = 0;

  const v4i gz = {0, 0, 0, 0};
#if TDM_SIX_ARGS
  const v8i gz8 = {0, 0, 0, 0, 0, 0, 0, 0};
  __builtin_amdgcn_tensor_load_to_lds(g0, g1, gz, gz, gz8, 0);
#else
  __builtin_amdgcn_tensor_load_to_lds(g0, g1, gz, gz, 0);
#endif
}
#endif  // USE_TDM

// ---------------------------------------------------------------- WMMA core
// Computes a 128x128 f32 tile of A (rowA0..+128) x B^T (rowB0..+128), where
// A is [*, K] bf16 row-major and B is [*, K] bf16 row-major (A.Bt GEMM).
// 256 threads = 8 waves; wave grid 4 (rows) x 2 (cols); each wave owns a
// 32x64 sub-tile = 2x4 WMMA 16x16 tiles. K must be a multiple of 32.
// Double-buffered LDS staging via the Tensor Data Mover (fallback: per-
// thread register staging); v_wmma_f32_16x16x32_bf16 inner product.
__device__ __forceinline__ void mma_tile_128x128(
    const __bf16* __restrict__ A, const __bf16* __restrict__ B,
    int K, long rowA0, long rowB0, v8f acc[2][4])
{
  __shared__ __attribute__((aligned(16))) __bf16 As[2][128][32];
  __shared__ __attribute__((aligned(16))) __bf16 Bs[2][128][32];

  const int tid    = threadIdx.x;
  const int lane   = tid & 31;
  const int lane16 = lane & 15;
  const int hi     = lane >> 4;       // which half of the wave
  const int wave   = tid >> 5;
  const int waveR  = wave & 3;        // 0..3 : 32-row strip
  const int waveC  = wave >> 2;       // 0..1 : 64-col strip

  v8f zero;
#pragma unroll
  for (int i = 0; i < 8; ++i) zero[i] = 0.0f;
#pragma unroll
  for (int rt = 0; rt < 2; ++rt)
#pragma unroll
    for (int ct = 0; ct < 4; ++ct) acc[rt][ct] = zero;

  const int nsteps = K >> 5;

#if USE_TDM
  // DMA engine fills whole 8KB tiles; only wave 0 issues (TDM ignores EXEC,
  // so issuing from every wave would duplicate the transfer).
  const __bf16* aTile = A + rowA0 * (long)K;
  const __bf16* bTile = B + rowB0 * (long)K;
  const unsigned asLds = (unsigned)(size_t)&As[0][0][0];
  const unsigned bsLds = (unsigned)(size_t)&Bs[0][0][0];

  if (wave == 0) {
    tdm_load_tile_128x32(aTile, asLds, K);
    tdm_load_tile_128x32(bTile, bsLds, K);
    __builtin_amdgcn_s_wait_tensorcnt(0);
  }
  __syncthreads();

  for (int s = 0; s < nsteps; ++s) {
    const int cur = s & 1;
    if (s + 1 < nsteps && wave == 0) {
      const unsigned boff = (unsigned)(cur ^ 1) * (128u * 32u * 2u);
      tdm_load_tile_128x32(aTile + (long)(s + 1) * 32, asLds + boff, K);
      tdm_load_tile_128x32(bTile + (long)(s + 1) * 32, bsLds + boff, K);
    }
#else
  // Fallback: per-thread register staging, each thread owns 16 bf16 (32B).
  const int lrow = tid >> 1;          // 0..127
  const int lk   = (tid & 1) * 16;    // 0 or 16
  const __bf16* aSrc = A + (rowA0 + lrow) * (long)K + lk;
  const __bf16* bSrc = B + (rowB0 + lrow) * (long)K + lk;

  v8bf ra0 = *(const v8bf*)(aSrc);
  v8bf ra1 = *(const v8bf*)(aSrc + 8);
  v8bf rb0 = *(const v8bf*)(bSrc);
  v8bf rb1 = *(const v8bf*)(bSrc + 8);
  *(v8bf*)&As[0][lrow][lk]     = ra0;
  *(v8bf*)&As[0][lrow][lk + 8] = ra1;
  *(v8bf*)&Bs[0][lrow][lk]     = rb0;
  *(v8bf*)&Bs[0][lrow][lk + 8] = rb1;
  __syncthreads();

  for (int s = 0; s < nsteps; ++s) {
    const int cur = s & 1;
    if (s + 1 < nsteps) {
      const __bf16* an = aSrc + (long)(s + 1) * 32;
      const __bf16* bn = bSrc + (long)(s + 1) * 32;
      ra0 = *(const v8bf*)(an);
      ra1 = *(const v8bf*)(an + 8);
      rb0 = *(const v8bf*)(bn);
      rb1 = *(const v8bf*)(bn + 8);
      if (s + 2 < nsteps) {   // lowers to global_prefetch_b8
        __builtin_prefetch(aSrc + (long)(s + 2) * 32, 0, 0);
        __builtin_prefetch(bSrc + (long)(s + 2) * 32, 0, 0);
      }
    }
#endif

    // A fragment, 16x32 bf16 per ISA layout:
    //   lanes 0-15 : M=lane, K 0..7 (v0-3) and 16..23 (v4-7)
    //   lanes16-31 : M=lane-16, K 8..15 and 24..31
    v16bf afr[2];
#pragma unroll
    for (int rt = 0; rt < 2; ++rt) {
      const __bf16* ap = &As[cur][waveR * 32 + rt * 16 + lane16][0];
      v8bf alo = *(const v8bf*)(ap + hi * 8);
      v8bf ahi = *(const v8bf*)(ap + 16 + hi * 8);
      afr[rt] = cat8(alo, ahi);
    }
    // B fragment, 32x16 bf16: lane = N; lanes 0-15 hold K 0..7 and 16..23,
    // lanes 16-31 hold K 8..15 and 24..31 (same order as A).
    v16bf bfr[4];
#pragma unroll
    for (int ct = 0; ct < 4; ++ct) {
      const __bf16* bp = &Bs[cur][waveC * 64 + ct * 16 + lane16][0];
      v8bf blo = *(const v8bf*)(bp + hi * 8);
      v8bf bhi = *(const v8bf*)(bp + 16 + hi * 8);
      bfr[ct] = cat8(blo, bhi);
    }

#pragma unroll
    for (int rt = 0; rt < 2; ++rt)
#pragma unroll
      for (int ct = 0; ct < 4; ++ct)
        acc[rt][ct] = __builtin_amdgcn_wmma_f32_16x16x32_bf16(
            false, afr[rt], false, bfr[ct], (short)0, acc[rt][ct],
            false, false);

#if USE_TDM
    if (s + 1 < nsteps) {
      if (wave == 0) __builtin_amdgcn_s_wait_tensorcnt(0);
      __syncthreads();
    }
  }
#else
    if (s + 1 < nsteps) {
      const int nxt = cur ^ 1;
      *(v8bf*)&As[nxt][lrow][lk]     = ra0;
      *(v8bf*)&As[nxt][lrow][lk + 8] = ra1;
      *(v8bf*)&Bs[nxt][lrow][lk]     = rb0;
      *(v8bf*)&Bs[nxt][lrow][lk + 8] = rb1;
      __syncthreads();
    }
  }
#endif
}

// ---------------------------------------------------------------- GEMM
// C = A . B^T (+bias).  Output bf16 (optionally transposed) or f32.
template <bool HAS_BIAS, bool TRANS_OUT, bool OUT_F32>
__global__ __launch_bounds__(256) void gemm_bf16_kernel(
    const __bf16* __restrict__ A, const __bf16* __restrict__ B,
    const float* __restrict__ bias, void* __restrict__ outp,
    int M, int N, int K)
{
  v8f acc[2][4];
  const long rowA0 = (long)blockIdx.y * 128;
  const long rowB0 = (long)blockIdx.x * 128;
  mma_tile_128x128(A, B, K, rowA0, rowB0, acc);

  const int tid = threadIdx.x;
  const int lane = tid & 31, lane16 = lane & 15, hi = lane >> 4;
  const int wave = tid >> 5, waveR = wave & 3, waveC = wave >> 2;
  const long rowBase = rowA0 + waveR * 32;
  const long colBase = rowB0 + waveC * 64;

#pragma unroll
  for (int rt = 0; rt < 2; ++rt) {
#pragma unroll
    for (int ct = 0; ct < 4; ++ct) {
      const long col = colBase + ct * 16 + lane16;
      float bv = 0.0f;
      if (HAS_BIAS) bv = bias[col];
#pragma unroll
      for (int e = 0; e < 8; ++e) {
        const long row = rowBase + rt * 16 + hi * 8 + e;  // C layout: M=e+8*hi
        const float vv = acc[rt][ct][e] + bv;
        if (OUT_F32) {
          ((float*)outp)[row * (long)N + col] = vv;
        } else if (TRANS_OUT) {
          ((__bf16*)outp)[col * (long)M + row] = (__bf16)vv;
        } else {
          ((__bf16*)outp)[row * (long)N + col] = (__bf16)vv;
        }
      }
    }
  }
}

// ---------------------------------------------------------------- scores
// S = 0.25 * q.k^T over a 128x128 tile; softmax is over axis 0 (rows i),
// i.e. per column j. MODE 0: column max (atomicMax on encoded uint).
// MODE 1: per-i-block partial sum of exp(S - m_j)   (deterministic).
// MODE 2: write attn[i,j] = exp(S - m_j) / Z_j as bf16.
template <int MODE>
__global__ __launch_bounds__(256) void scores_kernel(
    const __bf16* __restrict__ Q, const __bf16* __restrict__ Kmat,
    unsigned* __restrict__ colmax, float* __restrict__ partial,
    const float* __restrict__ colsum, __bf16* __restrict__ attn)
{
  v8f acc[2][4];
  const long i0 = (long)blockIdx.y * 128;
  const long j0 = (long)blockIdx.x * 128;
  mma_tile_128x128(Q, Kmat, HID, i0, j0, acc);

  const int tid = threadIdx.x;
  const int lane = tid & 31, lane16 = lane & 15, hi = lane >> 4;
  const int wave = tid >> 5, waveR = wave & 3, waveC = wave >> 2;

  __shared__ float red[4][128];

  if (MODE == 0) {
#pragma unroll
    for (int ct = 0; ct < 4; ++ct) {
      float m = -3.4e38f;
#pragma unroll
      for (int rt = 0; rt < 2; ++rt)
#pragma unroll
        for (int e = 0; e < 8; ++e) m = fmaxf(m, acc[rt][ct][e]);
      m = fmaxf(m, __shfl_xor(m, 16, 32));  // combine wave halves (same col)
      if (hi == 0) red[waveR][waveC * 64 + ct * 16 + lane16] = m;
    }
    __syncthreads();
    if (tid < 128) {
      float m = fmaxf(fmaxf(red[0][tid], red[1][tid]),
                      fmaxf(red[2][tid], red[3][tid]));
      atomicMax(&colmax[j0 + tid], enc_f32(0.25f * m));
    }
  } else if (MODE == 1) {
#pragma unroll
    for (int ct = 0; ct < 4; ++ct) {
      const long col = j0 + waveC * 64 + ct * 16 + lane16;
      const float m = dec_f32(colmax[col]);
      float ssum = 0.0f;
#pragma unroll
      for (int rt = 0; rt < 2; ++rt)
#pragma unroll
        for (int e = 0; e < 8; ++e)
          ssum += expf(0.25f * acc[rt][ct][e] - m);
      ssum += __shfl_xor(ssum, 16, 32);
      if (hi == 0) red[waveR][waveC * 64 + ct * 16 + lane16] = ssum;
    }
    __syncthreads();
    if (tid < 128) {
      const float s = red[0][tid] + red[1][tid] + red[2][tid] + red[3][tid];
      partial[(long)blockIdx.y * N_ROWS + j0 + tid] = s;  // no float atomics
    }
  } else {
#pragma unroll
    for (int ct = 0; ct < 4; ++ct) {
      const long col = j0 + waveC * 64 + ct * 16 + lane16;
      const float m  = dec_f32(colmax[col]);
      const float rz = 1.0f / colsum[col];
#pragma unroll
      for (int rt = 0; rt < 2; ++rt) {
#pragma unroll
        for (int e = 0; e < 8; ++e) {
          const long row = i0 + waveR * 32 + rt * 16 + hi * 8 + e;
          const float p = expf(0.25f * acc[rt][ct][e] - m) * rz;
          attn[row * (long)N_ROWS + col] = (__bf16)p;
        }
      }
    }
  }
}

// ---------------------------------------------------------------- misc

__global__ __launch_bounds__(256) void cvt_f32_to_bf16_kernel(
    const float* __restrict__ src, __bf16* __restrict__ dst, long n)
{
  const long i = ((long)blockIdx.x * 256 + threadIdx.x) * 4;
  if (i + 3 < n) {
    const float4 f = *(const float4*)(src + i);
    v4bf o;
    o[0] = (__bf16)f.x; o[1] = (__bf16)f.y;
    o[2] = (__bf16)f.z; o[3] = (__bf16)f.w;
    *(v4bf*)(dst + i) = o;
  }
}

__global__ __launch_bounds__(256) void init_stats_kernel(
    unsigned* __restrict__ colmax)
{
  const int i = blockIdx.x * 256 + threadIdx.x;
  if (i < N_ROWS) colmax[i] = enc_f32(-3.4e38f);
}

__global__ __launch_bounds__(256) void reduce_colsum_kernel(
    const float* __restrict__ partial, float* __restrict__ colsum)
{
  const int j = blockIdx.x * 256 + threadIdx.x;
  if (j < N_ROWS) {
    float s = 0.0f;
    for (int b = 0; b < N_ROWS / 128; ++b)   // fixed order -> deterministic
      s += partial[(long)b * N_ROWS + j];
    colsum[j] = s;
  }
}

__global__ __launch_bounds__(256) void logcosh_kernel(
    const float* __restrict__ O, float* __restrict__ out)
{
  const int row = blockIdx.x;
  const float* orow = O + (long)row * HID;
  float s = 0.0f;
  for (int h = threadIdx.x; h < HID; h += 256) {
    const float a = fabsf(orow[h]);
    s += a + log1pf(expf(-2.0f * a)) - 0.69314718055994531f;
  }
  __shared__ float red[256];
  red[threadIdx.x] = s;
  __syncthreads();
  for (int off = 128; off > 0; off >>= 1) {
    if (threadIdx.x < off) red[threadIdx.x] += red[threadIdx.x + off];
    __syncthreads();
  }
  if (threadIdx.x == 0) out[row] = red[0];
}

// ---------------------------------------------------------------- launch

extern "C" void kernel_launch(void* const* d_in, const int* in_sizes, int n_in,
                              void* d_out, int out_size, void* d_ws,
                              size_t ws_size, hipStream_t stream)
{
  (void)in_sizes; (void)n_in; (void)out_size; (void)ws_size;

  const float* x    = (const float*)d_in[0];
  const float* W_in = (const float*)d_in[1];
  const float* b_in = (const float*)d_in[2];
  const float* Wq   = (const float*)d_in[3];
  const float* Wk   = (const float*)d_in[4];
  const float* Wv   = (const float*)d_in[5];

  char* ws = (char*)d_ws;
  size_t off = 0;
  auto take = [&](size_t bytes) -> char* {
    char* p = ws + off;
    off += (bytes + 255) & ~(size_t)255;
    return p;
  };

  const size_t NB = (size_t)N_ROWS;
  __bf16*   xb     = (__bf16*)take(NB * HID * 2);        // x bf16, reused as q
  __bf16*   hb     = (__bf16*)take(NB * HID * 2);        // h bf16
  __bf16*   kb     = (__bf16*)take(NB * HID * 2);        // k bf16
  __bf16*   vtb    = (__bf16*)take(NB * HID * 2);        // v^T bf16 [HID, N]
  __bf16*   wb0    = (__bf16*)take((size_t)HID * HID * 2);
  __bf16*   wb1    = (__bf16*)take((size_t)HID * HID * 2);
  __bf16*   wb2    = (__bf16*)take((size_t)HID * HID * 2);
  __bf16*   wb3    = (__bf16*)take((size_t)HID * HID * 2);
  unsigned* colmax = (unsigned*)take(NB * 4);
  float*    colsum = (float*)take(NB * 4);
  float*    part   = (float*)take((NB / 128) * NB * 4);  // per-i-block sums
  float*    O      = (float*)take(NB * HID * 4);         // attn @ v, f32
  __bf16*   attn   = (__bf16*)take(NB * NB * 2);         // 128 MB bf16
  __bf16*   qb     = xb;                                 // alias (x is dead)

  // fp32 -> bf16 conversions
  cvt_f32_to_bf16_kernel<<<(N_ROWS * HID) / 1024, 256, 0, stream>>>(
      x, xb, (long)N_ROWS * HID);
  cvt_f32_to_bf16_kernel<<<(HID * HID) / 1024, 256, 0, stream>>>(
      W_in, wb0, (long)HID * HID);
  cvt_f32_to_bf16_kernel<<<(HID * HID) / 1024, 256, 0, stream>>>(
      Wq, wb1, (long)HID * HID);
  cvt_f32_to_bf16_kernel<<<(HID * HID) / 1024, 256, 0, stream>>>(
      Wk, wb2, (long)HID * HID);
  cvt_f32_to_bf16_kernel<<<(HID * HID) / 1024, 256, 0, stream>>>(
      Wv, wb3, (long)HID * HID);

  const dim3 g1(HID / 128, N_ROWS / 128);   // [M=8192, N=1024] GEMMs
  // h = x W_in^T + b
  gemm_bf16_kernel<true, false, false><<<g1, 256, 0, stream>>>(
      xb, wb0, b_in, hb, N_ROWS, HID, HID);
  // q, k (q overwrites xb region; A is hb, so no aliasing within a kernel)
  gemm_bf16_kernel<false, false, false><<<g1, 256, 0, stream>>>(
      hb, wb1, nullptr, qb, N_ROWS, HID, HID);
  gemm_bf16_kernel<false, false, false><<<g1, 256, 0, stream>>>(
      hb, wb2, nullptr, kb, N_ROWS, HID, HID);
  // v, written transposed so attn@v reuses the A.Bt kernel
  gemm_bf16_kernel<false, true, false><<<g1, 256, 0, stream>>>(
      hb, wb3, nullptr, vtb, N_ROWS, HID, HID);

  // column softmax stats + attn materialization
  init_stats_kernel<<<N_ROWS / 256, 256, 0, stream>>>(colmax);
  const dim3 gs(N_ROWS / 128, N_ROWS / 128);
  scores_kernel<0><<<gs, 256, 0, stream>>>(qb, kb, colmax, part, colsum, attn);
  scores_kernel<1><<<gs, 256, 0, stream>>>(qb, kb, colmax, part, colsum, attn);
  reduce_colsum_kernel<<<N_ROWS / 256, 256, 0, stream>>>(part, colsum);
  scores_kernel<2><<<gs, 256, 0, stream>>>(qb, kb, colmax, part, colsum, attn);

  // O = attn @ v   (A=attn [N,N] bf16, B=v^T [HID,N] bf16, K=8192, f32 out)
  gemm_bf16_kernel<false, false, true><<<g1, 256, 0, stream>>>(
      attn, vtb, nullptr, O, N_ROWS, HID, N_ROWS);

  // d_out[i] = sum_h logcosh(O[i,h])
  logcosh_kernel<<<N_ROWS, 256, 0, stream>>>(O, (float*)d_out);
}
